// GNNDiffPool_23759759081884
// MI455X (gfx1250) — matrix operand order repaired and checked by reading, not compiled, not run on hardware
//
#include <hip/hip_runtime.h>
#include <hip/hip_bf16.h>
#include <math.h>

// ---------------------------------------------------------------------------
// Problem constants (match reference setup_inputs)
// ---------------------------------------------------------------------------
#define B_G     64
#define N_NODES 512
#define DEG_E   32
#define NV      (B_G * N_NODES)          // 32768 nodes
#define NE      (B_G * N_NODES * DEG_E)  // 1,048,576 edges
#define IN_DIM  16
#define EMB_D   32
#define POOL_D  16
#define BN_EPS  1e-5f
#define EPS_LOG 1e-15f

typedef _Float16 v16h  __attribute__((ext_vector_type(16)));
typedef _Float16 v8h   __attribute__((ext_vector_type(8)));
typedef float    v8f   __attribute__((ext_vector_type(8)));
typedef unsigned int u32x4 __attribute__((ext_vector_type(4)));
typedef int          i32x8 __attribute__((ext_vector_type(8)));
typedef int          i32x4 __attribute__((ext_vector_type(4)));

#define TSTRIDE 40   // LDS row stride in halfs: 80B -> 16B aligned vectors, bank-spread

// ---------------------------------------------------------------------------
// TDM: DMA a 2D f32 tile (tile_d0 x tile_d1, row stride stride0 elements) from
// global memory into contiguous LDS. Descriptor per CDNA5 ISA.
// OOB reads (beyond tensor_d0/d1) return zero -> free K-tail padding.
// Completion tracked on TENSORcnt.
// ---------------------------------------------------------------------------
__device__ __forceinline__ void tdm_load_2d_f32(unsigned lds_addr, const float* gaddr,
                                                unsigned tensor_d0, unsigned tensor_d1,
                                                unsigned long long stride0_elems,
                                                unsigned tile_d0, unsigned tile_d1)
{
    unsigned long long ga = (unsigned long long)gaddr;
    u32x4 g0;
    g0[0] = 1u;                                            // count=1, user descriptor
    g0[1] = lds_addr;                                      // LDS byte address
    g0[2] = (unsigned)(ga & 0xFFFFFFFFu);                  // global_addr[31:0]
    g0[3] = (unsigned)((ga >> 32) & 0x01FFFFFFu)           // global_addr[56:32]
          | (2u << 30);                                    // type = 2 ("image")
    i32x8 g1;
    g1[0] = (int)(2u << 16);                               // data_size = 4B
    g1[1] = (int)((tensor_d0 & 0xFFFFu) << 16);            // tensor_dim0[15:0]  @ [63:48]
    g1[2] = (int)(((tensor_d0 >> 16) & 0xFFFFu)            // tensor_dim0[31:16] @ [79:64]
                | ((tensor_d1 & 0xFFFFu) << 16));          // tensor_dim1[15:0]  @ [95:80]
    g1[3] = (int)(((tensor_d1 >> 16) & 0xFFFFu)            // tensor_dim1[31:16]
                | ((tile_d0 & 0xFFFFu) << 16));            // tile_dim0 @ [127:112]
    g1[4] = (int)(tile_d1 & 0xFFFFu);                      // tile_dim1; tile_dim2=0
    g1[5] = (int)(unsigned)(stride0_elems & 0xFFFFFFFFu);  // tensor_dim0_stride[31:0]
    g1[6] = (int)(unsigned)((stride0_elems >> 32) & 0xFFFFu); // stride[47:32]; dim1_stride lo16=0
    g1[7] = 0;
    i32x4 z4 = {};
    i32x8 z8 = {};
    // 6-arg form (clang-23 / therock-10.0 headers): groups 2,3 unused, extra group, cpol=0
    __builtin_amdgcn_tensor_load_to_lds(g0, g1, z4, z4, z8, 0);
}

// ---------------------------------------------------------------------------
// LDS-staged batched GEMM on the CDNA5 WMMA path: C = A @ B (fp32 in/out,
// v_wmma_f32_16x16x32_f16, f32 accumulate). One wave per 16x16 C tile.
//
// TRANS == 0:  A is [M x K] row-major, M % 16 == 0, K % 16 == 0.
//              A tile is DMA'd global->LDS by the Tensor Data Mover.
// TRANS == 1:  A = S^T where S is [K x 8] row-major *contiguous* (lda ignored),
//              M == 8, K % 32 == 0.
// B is [K x N] row-major; N may be ragged (8/16/32 here).
// Fragment lane layouts per CDNA5 ISA §7.12.2.
// ---------------------------------------------------------------------------
template <int TRANS>
__global__ __launch_bounds__(32)
void wmma_gemm_k(const float* __restrict__ A, const float* __restrict__ Bm,
                 float* __restrict__ C,
                 int M, int K, int N, int lda, int ldb, int ldc,
                 long long sA, long long sB, long long sC)
{
    __shared__ float    AsF[16 * 32];       // f32 A tile (TDM destination), [m][k]
    __shared__ _Float16 As[16 * TSTRIDE];   // f16 A tile for TRANS path, [m][k]
    __shared__ _Float16 BsT[16 * TSTRIDE];  // f16 B tile transposed, [n][k]

    const float* Ab = A  + (long long)blockIdx.z * sA;
    const float* Bb = Bm + (long long)blockIdx.z * sB;
    float*       Cb = C  + (long long)blockIdx.z * sC;

    const int m0   = blockIdx.x * 16;
    const int n0   = blockIdx.y * 16;
    const int lane = threadIdx.x;
    const int grp  = lane >> 4;          // 0: lanes 0-15, 1: lanes 16-31
    const int l15  = lane & 15;
    const int avail = N - n0;            // valid B/C columns in this tile (8 or 16 here)
    const unsigned ldsA = (unsigned)(unsigned long long)(void*)&AsF[0];

    if (TRANS) {
        // A-tile rows 8..15 are permanent zero padding (M == 8)
        for (int i = lane; i < 16 * TSTRIDE; i += 32) As[i] = (_Float16)0.0f;
        __syncthreads();
    }

    v8f acc = {};
    for (int k0 = 0; k0 < K; k0 += 32) {
        // ---- stage A tile (16 x 32) ----
        if (TRANS) {
            // source: S rows k0+lane (8 contiguous floats each); write transposed
            const float4* Sr = (const float4*)(Ab + (long long)(k0 + lane) * 8);
            float4 f0 = Sr[0], f1 = Sr[1];
            As[0 * TSTRIDE + lane] = (_Float16)f0.x;
            As[1 * TSTRIDE + lane] = (_Float16)f0.y;
            As[2 * TSTRIDE + lane] = (_Float16)f0.z;
            As[3 * TSTRIDE + lane] = (_Float16)f0.w;
            As[4 * TSTRIDE + lane] = (_Float16)f1.x;
            As[5 * TSTRIDE + lane] = (_Float16)f1.y;
            As[6 * TSTRIDE + lane] = (_Float16)f1.z;
            As[7 * TSTRIDE + lane] = (_Float16)f1.w;
        } else {
            // Tensor Data Mover: 16 rows x 32 f32 at row stride lda -> contiguous LDS.
            // OOB (k beyond K) reads return zero -> handles K==16 tail.
            tdm_load_2d_f32(ldsA, Ab + (long long)m0 * lda + k0,
                            (unsigned)(K - k0), 16u,
                            (unsigned long long)lda, 32u, 16u);
        }

        // ---- stage B tile (32 x 16) as f16, transposed to [n][k] ----
        {
            const int kr = k0 + lane;           // each lane one B row
            _Float16 hv[16];
            if (kr < K) {
                const float* Br = Bb + (long long)kr * ldb + n0;
                if (avail >= 16) {
                    const float4* Bv = (const float4*)Br;
                    float4 f0 = Bv[0], f1 = Bv[1], f2 = Bv[2], f3 = Bv[3];
                    hv[0]=(_Float16)f0.x; hv[1]=(_Float16)f0.y; hv[2]=(_Float16)f0.z; hv[3]=(_Float16)f0.w;
                    hv[4]=(_Float16)f1.x; hv[5]=(_Float16)f1.y; hv[6]=(_Float16)f1.z; hv[7]=(_Float16)f1.w;
                    hv[8]=(_Float16)f2.x; hv[9]=(_Float16)f2.y; hv[10]=(_Float16)f2.z; hv[11]=(_Float16)f2.w;
                    hv[12]=(_Float16)f3.x; hv[13]=(_Float16)f3.y; hv[14]=(_Float16)f3.z; hv[15]=(_Float16)f3.w;
                } else {        // avail == 8 in this network
                    const float4* Bv = (const float4*)Br;
                    float4 f0 = Bv[0], f1 = Bv[1];
                    hv[0]=(_Float16)f0.x; hv[1]=(_Float16)f0.y; hv[2]=(_Float16)f0.z; hv[3]=(_Float16)f0.w;
                    hv[4]=(_Float16)f1.x; hv[5]=(_Float16)f1.y; hv[6]=(_Float16)f1.z; hv[7]=(_Float16)f1.w;
#pragma unroll
                    for (int j = 8; j < 16; ++j) hv[j] = (_Float16)0.0f;
                }
            } else {
#pragma unroll
                for (int j = 0; j < 16; ++j) hv[j] = (_Float16)0.0f;
            }
#pragma unroll
            for (int j = 0; j < 16; ++j) BsT[j * TSTRIDE + lane] = hv[j];
        }
        if (!TRANS) __builtin_amdgcn_s_wait_tensorcnt(0);   // A tile DMA complete
        __syncthreads();

        // ---- build fragments from LDS (aligned 16B ds loads) ----
        v16h a, b;
        if (TRANS) {
            // A: lane -> row m = l15; K halfs {grp*8+0..7, grp*8+16..23}
            v8h alo = *(v8h*)&As[l15 * TSTRIDE + grp * 8];
            v8h ahi = *(v8h*)&As[l15 * TSTRIDE + grp * 8 + 16];
#pragma unroll
            for (int i = 0; i < 8; ++i) { a[i] = alo[i]; a[i + 8] = ahi[i]; }
        } else {
            // A tile is f32 in LDS; convert while packing the fragment
            const float* Ar = &AsF[l15 * 32 + grp * 8];
            float4 q0 = *(const float4*)(Ar);
            float4 q1 = *(const float4*)(Ar + 4);
            float4 q2 = *(const float4*)(Ar + 16);
            float4 q3 = *(const float4*)(Ar + 20);
            a[0]=(_Float16)q0.x; a[1]=(_Float16)q0.y; a[2]=(_Float16)q0.z; a[3]=(_Float16)q0.w;
            a[4]=(_Float16)q1.x; a[5]=(_Float16)q1.y; a[6]=(_Float16)q1.z; a[7]=(_Float16)q1.w;
            a[8]=(_Float16)q2.x; a[9]=(_Float16)q2.y; a[10]=(_Float16)q2.z; a[11]=(_Float16)q2.w;
            a[12]=(_Float16)q3.x; a[13]=(_Float16)q3.y; a[14]=(_Float16)q3.z; a[15]=(_Float16)q3.w;
        }
        // B: lane -> col n = l15; K halfs {grp*8+0..7, grp*8+16..23}
        v8h blo = *(v8h*)&BsT[l15 * TSTRIDE + grp * 8];
        v8h bhi = *(v8h*)&BsT[l15 * TSTRIDE + grp * 8 + 16];
#pragma unroll
        for (int i = 0; i < 8; ++i) { b[i] = blo[i]; b[i + 8] = bhi[i]; }

        acc = __builtin_amdgcn_wmma_f32_16x16x32_f16(
                false, a, false, b, (short)0, acc, false, false);
        __syncthreads();
    }

    // ---- store C: VGPR r, lane l -> (M = m0 + r + 8*grp, N = n0 + l15) ----
    const int ncol = n0 + l15;
    if (ncol < N) {
#pragma unroll
        for (int r = 0; r < 8; ++r) {
            int mr = m0 + r + 8 * grp;
            if (mr < M) Cb[(long long)mr * ldc + ncol] = acc[r];
        }
    }
}

// ---------------------------------------------------------------------------
// Elementwise / scatter / reduction helpers
// ---------------------------------------------------------------------------
__global__ void fill_kernel(float* p, float v, long long n)
{
    long long t = (long long)blockIdx.x * blockDim.x + threadIdx.x;
    if (t < n) p[t] = v;
}

__global__ void edge_weight_kernel(const float* __restrict__ pos,
                                   const int* __restrict__ src,
                                   const int* __restrict__ dst,
                                   float* __restrict__ w, long long E)
{
    long long e = (long long)blockIdx.x * blockDim.x + threadIdx.x;
    if (e >= E) return;
    int s = src[e], d = dst[e];
    float dx = pos[3 * s + 0] - pos[3 * d + 0];
    float dy = pos[3 * s + 1] - pos[3 * d + 1];
    float dz = pos[3 * s + 2] - pos[3 * d + 2];
    w[e] = __expf(-0.5f * (dx * dx + dy * dy + dz * dz));
}

__global__ void deg_scatter_kernel(const int* __restrict__ dst,
                                   const float* __restrict__ w,
                                   float* __restrict__ deg, long long E)
{
    long long e = (long long)blockIdx.x * blockDim.x + threadIdx.x;
    if (e >= E) return;
    atomicAdd(&deg[dst[e]], w[e]);
}

__global__ void rsqrt_kernel(const float* __restrict__ deg, float* __restrict__ nrm, int n)
{
    int i = blockIdx.x * blockDim.x + threadIdx.x;
    if (i < n) nrm[i] = rsqrtf(deg[i]);
}

// scatter: agg[dst,f] += norm[src]*w*norm[dst] * h[src,f]
__global__ void edge_scatter_kernel(const int* __restrict__ src, const int* __restrict__ dst,
                                    const float* __restrict__ w, const float* __restrict__ nrm,
                                    const float* __restrict__ h, float* __restrict__ agg,
                                    int dout, long long total)
{
    long long t = (long long)blockIdx.x * blockDim.x + threadIdx.x;
    if (t >= total) return;
    int e = (int)(t / dout), f = (int)(t % dout);
    int s = src[e], d = dst[e];
    float coef = nrm[s] * w[e] * nrm[d];
    atomicAdd(&agg[(long long)d * dout + f], coef * h[(long long)s * dout + f]);
}

// out = agg + norm^2 * h + bias  (+ optional ReLU)
__global__ void conv_combine_kernel(const float* __restrict__ agg, const float* __restrict__ h,
                                    const float* __restrict__ nrm, const float* __restrict__ bias,
                                    float* __restrict__ out, int dout, long long total, int relu)
{
    long long t = (long long)blockIdx.x * blockDim.x + threadIdx.x;
    if (t >= total) return;
    int i = (int)(t / dout), f = (int)(t % dout);
    float nn = nrm[i] * nrm[i];
    float v = agg[t] + nn * h[t] + bias[f];
    out[t] = relu ? fmaxf(v, 0.0f) : v;
}

__global__ void adj_scatter_kernel(const int* __restrict__ src, const int* __restrict__ dst,
                                   const float* __restrict__ w, float* __restrict__ adj,
                                   long long E)
{
    long long e = (long long)blockIdx.x * blockDim.x + threadIdx.x;
    if (e >= E) return;
    int s = src[e], d = dst[e];
    int g = s / N_NODES;
    atomicAdd(&adj[((long long)g * N_NODES + (s % N_NODES)) * N_NODES + (d % N_NODES)], w[e]);
}

// BatchNorm (training) stats: per feature over all rows
__global__ void bn_stats_kernel(const float* __restrict__ x, int rows, int d,
                                float* __restrict__ mean, float* __restrict__ var)
{
    __shared__ float s1[256], s2[256];
    int f = blockIdx.x;
    float a = 0.f, b = 0.f;
    for (int r = threadIdx.x; r < rows; r += blockDim.x) {
        float v = x[(long long)r * d + f];
        a += v; b += v * v;
    }
    s1[threadIdx.x] = a; s2[threadIdx.x] = b;
    __syncthreads();
    for (int o = 128; o > 0; o >>= 1) {
        if (threadIdx.x < o) { s1[threadIdx.x] += s1[threadIdx.x + o];
                               s2[threadIdx.x] += s2[threadIdx.x + o]; }
        __syncthreads();
    }
    if (threadIdx.x == 0) {
        float m = s1[0] / rows;
        mean[f] = m;
        var[f]  = s2[0] / rows - m * m;   // population var (ddof=0)
    }
}

__global__ void bn_apply_relu_kernel(float* __restrict__ x, const float* __restrict__ g,
                                     const float* __restrict__ be, const float* __restrict__ mean,
                                     const float* __restrict__ var, int d, long long total)
{
    long long t = (long long)blockIdx.x * blockDim.x + threadIdx.x;
    if (t >= total) return;
    int f = (int)(t % d);
    float v = x[t];
    v = g[f] * (v - mean[f]) * rsqrtf(var[f] + BN_EPS) + be[f];
    x[t] = fmaxf(v, 0.0f);
}

__global__ void softmax_rows_kernel(const float* __restrict__ sin, float* __restrict__ sout,
                                    int rows, int C)
{
    int r = blockIdx.x * blockDim.x + threadIdx.x;
    if (r >= rows) return;
    float mx = -1e30f;
    for (int c = 0; c < C; ++c) mx = fmaxf(mx, sin[(long long)r * C + c]);
    float sum = 0.f;
    for (int c = 0; c < C; ++c) {
        float e = __expf(sin[(long long)r * C + c] - mx);
        sout[(long long)r * C + c] = e; sum += e;
    }
    float inv = 1.0f / sum;
    for (int c = 0; c < C; ++c) sout[(long long)r * C + c] *= inv;
}

// link-loss sum of squares for level 0 (C fixed to 8)
__global__ void link0_kernel(const float* __restrict__ adj, const float* __restrict__ s,
                             float* __restrict__ acc, long long total)
{
    __shared__ float red[256];
    long long t = (long long)blockIdx.x * blockDim.x + threadIdx.x;
    float lp = 0.f;
    if (t < total) {
        int b = (int)(t / ((long long)N_NODES * N_NODES));
        long long r = t % ((long long)N_NODES * N_NODES);
        int i = (int)(r / N_NODES), j = (int)(r % N_NODES);
        const float4* si = (const float4*)(s + ((long long)b * N_NODES + i) * 8);
        const float4* sj = (const float4*)(s + ((long long)b * N_NODES + j) * 8);
        float4 a0 = si[0], a1 = si[1], b0 = sj[0], b1 = sj[1];
        float dot = a0.x * b0.x + a0.y * b0.y + a0.z * b0.z + a0.w * b0.w
                  + a1.x * b1.x + a1.y * b1.y + a1.z * b1.z + a1.w * b1.w;
        float df = adj[t] - dot;
        lp = df * df;
    }
    red[threadIdx.x] = lp;
    __syncthreads();
    for (int o = 128; o > 0; o >>= 1) {
        if (threadIdx.x < o) red[threadIdx.x] += red[threadIdx.x + o];
        __syncthreads();
    }
    if (threadIdx.x == 0) atomicAdd(acc, red[0]);
}

__global__ void entropy_kernel(const float* __restrict__ s, float* __restrict__ acc,
                               int rows, int C)
{
    __shared__ float red[256];
    int r = blockIdx.x * blockDim.x + threadIdx.x;
    float e = 0.f;
    if (r < rows)
        for (int c = 0; c < C; ++c) {
            float v = s[(long long)r * C + c];
            e += -v * logf(v + EPS_LOG);
        }
    red[threadIdx.x] = e;
    __syncthreads();
    for (int o = 128; o > 0; o >>= 1) {
        if (threadIdx.x < o) red[threadIdx.x] += red[threadIdx.x + o];
        __syncthreads();
    }
    if (threadIdx.x == 0) atomicAdd(acc, red[0]);
}

// Dense GCN conv for the tiny levels (n<=8): one block per graph.
__global__ __launch_bounds__(128)
void dense_conv_kernel(const float* __restrict__ x, const float* __restrict__ adj,
                       const float* __restrict__ W, const float* __restrict__ bias,
                       float* __restrict__ out, int n, int din, int dout, int relu)
{
    __shared__ float h[8 * 32];
    __shared__ float nr[8];
    int b = blockIdx.x, tid = threadIdx.x;
    const float* xb = x + (long long)b * n * din;
    const float* ab = adj + (long long)b * n * n;
    float* ob = out + (long long)b * n * dout;

    for (int idx = tid; idx < n * dout; idx += blockDim.x) {
        int i = idx / dout, d = idx % dout;
        float a = 0.f;
        for (int k = 0; k < din; ++k) a += xb[i * din + k] * W[k * dout + d];
        h[idx] = a;
    }
    __syncthreads();
    if (tid < n) {
        float sm = 1.0f;                      // self loop
        for (int j = 0; j < n; ++j) sm += ab[tid * n + j];
        nr[tid] = rsqrtf(sm);
    }
    __syncthreads();
    for (int idx = tid; idx < n * dout; idx += blockDim.x) {
        int i = idx / dout, d = idx % dout;
        float a = 0.f;
        for (int j = 0; j < n; ++j) {
            float aij = ab[i * n + j] + ((i == j) ? 1.0f : 0.0f);
            a += nr[i] * aij * nr[j] * h[j * dout + d];
        }
        float v = a + bias[d];
        ob[idx] = relu ? fmaxf(v, 0.0f) : v;
    }
}

// DiffPool for the tiny levels: one block per graph, 128 threads.
__global__ __launch_bounds__(128)
void diffpool_small_kernel(const float* __restrict__ x, const float* __restrict__ adj,
                           const float* __restrict__ s, float* __restrict__ xo,
                           float* __restrict__ ao, float* __restrict__ linkAcc,
                           float* __restrict__ entAcc, int n, int C, int d)
{
    __shared__ float t[8 * 8];
    __shared__ float red[128];
    int b = blockIdx.x, tid = threadIdx.x;
    const float* xb = x + (long long)b * n * d;
    const float* ab = adj + (long long)b * n * n;
    const float* sb = s + (long long)b * n * C;

    for (int idx = tid; idx < C * d; idx += blockDim.x) {           // Sᵀ X
        int c = idx / d, dd = idx % d;
        float a = 0.f;
        for (int i = 0; i < n; ++i) a += sb[i * C + c] * xb[i * d + dd];
        xo[(long long)b * C * d + idx] = a;
    }
    for (int idx = tid; idx < n * C; idx += blockDim.x) {           // A S
        int i = idx / C, c = idx % C;
        float a = 0.f;
        for (int j = 0; j < n; ++j) a += ab[i * n + j] * sb[j * C + c];
        t[idx] = a;
    }
    __syncthreads();
    for (int idx = tid; idx < C * C; idx += blockDim.x) {           // Sᵀ (A S)
        int c = idx / C, k = idx % C;
        float a = 0.f;
        for (int i = 0; i < n; ++i) a += sb[i * C + c] * t[i * C + k];
        ao[(long long)b * C * C + idx] = a;
    }
    float lp = 0.f;                                                 // ||A - SSᵀ||²
    for (int idx = tid; idx < n * n; idx += blockDim.x) {
        int i = idx / n, j = idx % n;
        float dot = 0.f;
        for (int c = 0; c < C; ++c) dot += sb[i * C + c] * sb[j * C + c];
        float df = ab[idx] - dot;
        lp += df * df;
    }
    float ep = 0.f;                                                 // entropy
    for (int i = tid; i < n; i += blockDim.x)
        for (int c = 0; c < C; ++c) {
            float v = sb[i * C + c];
            ep += -v * logf(v + EPS_LOG);
        }
    red[tid] = lp; __syncthreads();
    for (int o = 64; o > 0; o >>= 1) { if (tid < o) red[tid] += red[tid + o]; __syncthreads(); }
    if (tid == 0) atomicAdd(linkAcc, red[0]);
    __syncthreads();
    red[tid] = ep; __syncthreads();
    for (int o = 64; o > 0; o >>= 1) { if (tid < o) red[tid] += red[tid + o]; __syncthreads(); }
    if (tid == 0) atomicAdd(entAcc, red[0]);
}

// Head: y = sigmoid(mean_c(x) @ W + b), plus final loss scalars.
__global__ void finalize_kernel(const float* __restrict__ xf, const float* __restrict__ mlpW,
                                const float* __restrict__ mlpb, const float* __restrict__ linkRaw,
                                const float* __restrict__ entRaw, float* __restrict__ out)
{
    int b = blockIdx.x * blockDim.x + threadIdx.x;
    if (b < B_G) {
        float acc = 0.f;
        for (int d = 0; d < EMB_D; ++d) {
            float xm = 0.5f * (xf[(long long)b * 2 * EMB_D + d] +
                               xf[(long long)b * 2 * EMB_D + EMB_D + d]);
            acc += xm * mlpW[d];
        }
        out[b] = 1.0f / (1.0f + __expf(-(acc + mlpb[0])));
    }
    if (b == 0) {
        float l = sqrtf(linkRaw[0]) / ((float)B_G * N_NODES * N_NODES)
                + sqrtf(linkRaw[1]) / ((float)B_G * 8 * 8)
                + sqrtf(linkRaw[2]) / ((float)B_G * 4 * 4);
        out[B_G]     = l;
        out[B_G + 1] = entRaw[0] / ((float)B_G * N_NODES)
                     + entRaw[1] / ((float)B_G * 8)
                     + entRaw[2] / ((float)B_G * 4);
    }
}

// ---------------------------------------------------------------------------
// Host orchestration
// ---------------------------------------------------------------------------
extern "C" void kernel_launch(void* const* d_in, const int* in_sizes, int n_in,
                              void* d_out, int out_size, void* d_ws, size_t ws_size,
                              hipStream_t stream)
{
    (void)in_sizes; (void)n_in; (void)out_size; (void)ws_size;

    // JAX pytree leaf order (dict keys sorted):
    //  0:x 1:pos 2:edge_index 3:batch 4:num_graphs
    //  params: 'emb'[i] {W0..3, b0..3, beta0..2, gamma0..2} -> 14 leaves each (i=0..2),
    //          'mlp_W', 'mlp_b', 'pool'[i] same 14-leaf layout (i=0..2)
    auto F = [&](int i) { return (const float*)d_in[i]; };
    const float* x_in  = F(0);
    const float* pos   = F(1);
    const int*   src   = (const int*)d_in[2];
    const int*   dst   = src + NE;
    const int EMB_BASE[3]  = {5, 19, 33};
    const int MLP_W = 47, MLP_B = 48;
    const int POOL_BASE[3] = {49, 63, 77};
    // within a stack base: W at +0..3, b at +4..7, beta at +8..10, gamma at +11..13

    // ---- workspace carve ----
    char* wp = (char*)d_ws;
    auto carve = [&](size_t bytes) {
        void* r = (void*)wp;
        wp += (bytes + 255) & ~(size_t)255;
        return r;
    };
    float* w_e    = (float*)carve((size_t)NE * 4);
    float* deg    = (float*)carve((size_t)NV * 4);
    float* nrm    = (float*)carve((size_t)NV * 4);
    float* adj0   = (float*)carve((size_t)B_G * N_NODES * N_NODES * 4);   // 67 MB
    float* t_h    = (float*)carve((size_t)NV * 32 * 4);
    float* t_agg  = (float*)carve((size_t)NV * 32 * 4);
    float* t_x    = (float*)carve((size_t)NV * 32 * 4);
    float* s_sp   = (float*)carve((size_t)NV * 8 * 4);
    float* h_emb  = (float*)carve((size_t)NV * 32 * 4);
    float* s_soft = (float*)carve((size_t)NV * 8 * 4);
    float* as_b   = (float*)carve((size_t)NV * 8 * 4);
    float* meanb  = (float*)carve(32 * 4);
    float* varb   = (float*)carve(32 * 4);
    float* lraw   = (float*)carve(3 * 4);
    float* eraw   = (float*)carve(3 * 4);
    float* xd1    = (float*)carve((size_t)B_G * 8 * 32 * 4);
    float* adj1   = (float*)carve((size_t)B_G * 8 * 8 * 4);
    float* tmpd   = (float*)carve((size_t)B_G * 8 * 32 * 4);
    float* s_raw  = (float*)carve((size_t)B_G * 8 * 8 * 4);
    float* s_d    = (float*)carve((size_t)B_G * 8 * 8 * 4);
    float* x_emb  = (float*)carve((size_t)B_G * 8 * 32 * 4);
    float* xd2    = (float*)carve((size_t)B_G * 8 * 32 * 4);
    float* adj2   = (float*)carve((size_t)B_G * 8 * 8 * 4);
    float* xd3    = (float*)carve((size_t)B_G * 8 * 32 * 4);
    float* adj3   = (float*)carve((size_t)B_G * 8 * 8 * 4);

    auto blocks = [](long long n, int bs) { return (unsigned)((n + bs - 1) / bs); };

    auto fill = [&](float* p, float v, long long n) {
        fill_kernel<<<blocks(n, 256), 256, 0, stream>>>(p, v, n);
    };
    // C = A @ B, A row-major [MxK] (M%16==0, K%16==0)
    auto gemm_nt = [&](const float* A, const float* Bm, float* C, int M, int K, int N,
                       int lda, int ldb, int ldc, long long sA, long long sB, long long sC,
                       int batch) {
        dim3 g(M / 16, (N + 15) / 16, batch);
        wmma_gemm_k<0><<<g, 32, 0, stream>>>(A, Bm, C, M, K, N, lda, ldb, ldc, sA, sB, sC);
    };
    // C = S^T @ B, S contiguous [Kx8] (K%32==0), M=8
    auto gemm_t8 = [&](const float* S, const float* Bm, float* C, int K, int N,
                       int ldb, int ldc, long long sA, long long sB, long long sC,
                       int batch) {
        dim3 g(1, (N + 15) / 16, batch);
        wmma_gemm_k<1><<<g, 32, 0, stream>>>(S, Bm, C, 8, K, N, 8, ldb, ldc, sA, sB, sC);
    };
    auto sparse_conv = [&](const float* xin, int din, int dout, const float* W,
                           const float* bias, float* xout, int relu) {
        gemm_nt(xin, W, t_h, NV, din, dout, din, dout, dout, 0, 0, 0, 1);
        fill(t_agg, 0.0f, (long long)NV * dout);
        long long te = (long long)NE * dout;
        edge_scatter_kernel<<<blocks(te, 256), 256, 0, stream>>>(src, dst, w_e, nrm,
                                                                 t_h, t_agg, dout, te);
        long long tn = (long long)NV * dout;
        conv_combine_kernel<<<blocks(tn, 256), 256, 0, stream>>>(t_agg, t_h, nrm, bias,
                                                                 xout, dout, tn, relu);
    };
    auto bn_relu = [&](float* xio, int rows, int d, const float* g, const float* be) {
        bn_stats_kernel<<<d, 256, 0, stream>>>(xio, rows, d, meanb, varb);
        long long tn = (long long)rows * d;
        bn_apply_relu_kernel<<<blocks(tn, 256), 256, 0, stream>>>(xio, g, be, meanb, varb, d, tn);
    };
    auto sparse_stack = [&](int base, int din, int dh, int dout, float* xout, int out_relu) {
        sparse_conv(x_in, din, dh, F(base + 0), F(base + 4), t_x, 0);
        bn_relu(t_x, NV, dh, F(base + 11), F(base + 8));
        sparse_conv(t_x, dh, dh, F(base + 1), F(base + 5), t_x, 0);
        bn_relu(t_x, NV, dh, F(base + 12), F(base + 9));
        sparse_conv(t_x, dh, dh, F(base + 2), F(base + 6), t_x, 0);
        bn_relu(t_x, NV, dh, F(base + 13), F(base + 10));
        sparse_conv(t_x, dh, dout, F(base + 3), F(base + 7), xout, out_relu);
    };
    auto dense_stack = [&](const float* xin, const float* adjc, int n, int din, int dh,
                           int dout, int base, float* xout, int out_relu) {
        dense_conv_kernel<<<B_G, 128, 0, stream>>>(xin, adjc, F(base + 0), F(base + 4),
                                                   tmpd, n, din, dh, 0);
        bn_relu(tmpd, B_G * n, dh, F(base + 11), F(base + 8));
        dense_conv_kernel<<<B_G, 128, 0, stream>>>(tmpd, adjc, F(base + 1), F(base + 5),
                                                   tmpd, n, dh, dh, 0);
        bn_relu(tmpd, B_G * n, dh, F(base + 12), F(base + 9));
        dense_conv_kernel<<<B_G, 128, 0, stream>>>(tmpd, adjc, F(base + 2), F(base + 6),
                                                   tmpd, n, dh, dh, 0);
        bn_relu(tmpd, B_G * n, dh, F(base + 13), F(base + 10));
        dense_conv_kernel<<<B_G, 128, 0, stream>>>(tmpd, adjc, F(base + 3), F(base + 7),
                                                   xout, n, dh, dout, out_relu);
    };

    // ---- level 0: sparse ----
    edge_weight_kernel<<<blocks(NE, 256), 256, 0, stream>>>(pos, src, dst, w_e, NE);
    fill(deg, 1.0f, NV);
    deg_scatter_kernel<<<blocks(NE, 256), 256, 0, stream>>>(dst, w_e, deg, NE);
    rsqrt_kernel<<<blocks(NV, 256), 256, 0, stream>>>(deg, nrm, NV);
    fill(lraw, 0.0f, 3);
    fill(eraw, 0.0f, 3);

    sparse_stack(POOL_BASE[0], IN_DIM, POOL_D, 8, s_sp, 0);     // pool -> [NV,8]
    sparse_stack(EMB_BASE[0],  IN_DIM, EMB_D, EMB_D, h_emb, 1); // emb  -> [NV,32]

    fill(adj0, 0.0f, (long long)B_G * N_NODES * N_NODES);
    adj_scatter_kernel<<<blocks(NE, 256), 256, 0, stream>>>(src, dst, w_e, adj0, NE);

    softmax_rows_kernel<<<blocks(NV, 256), 256, 0, stream>>>(s_sp, s_soft, NV, 8);

    // xd1 = Sᵀ X        [64 x (8x32)],  K=512
    gemm_t8(s_soft, h_emb, xd1, N_NODES, EMB_D, EMB_D, EMB_D,
            (long long)N_NODES * 8, (long long)N_NODES * EMB_D, 8 * EMB_D, B_G);
    // AS = A S          [64 x (512x8)], K=512
    gemm_nt(adj0, s_soft, as_b, N_NODES, N_NODES, 8, N_NODES, 8, 8,
            (long long)N_NODES * N_NODES, (long long)N_NODES * 8, (long long)N_NODES * 8, B_G);
    // adj1 = Sᵀ (A S)   [64 x (8x8)],   K=512
    gemm_t8(s_soft, as_b, adj1, N_NODES, 8, 8, 8,
            (long long)N_NODES * 8, (long long)N_NODES * 8, 8 * 8, B_G);

    long long tot0 = (long long)B_G * N_NODES * N_NODES;
    link0_kernel<<<blocks(tot0, 256), 256, 0, stream>>>(adj0, s_soft, &lraw[0], tot0);
    entropy_kernel<<<blocks(NV, 256), 256, 0, stream>>>(s_soft, &eraw[0], NV, 8);

    // ---- level 1: dense, n=8 -> C=4 ----
    dense_stack(xd1, adj1, 8, EMB_D, POOL_D, 4, POOL_BASE[1], s_raw, 0);
    dense_stack(xd1, adj1, 8, EMB_D, EMB_D, EMB_D, EMB_BASE[1], x_emb, 1);
    softmax_rows_kernel<<<blocks(B_G * 8, 256), 256, 0, stream>>>(s_raw, s_d, B_G * 8, 4);
    diffpool_small_kernel<<<B_G, 128, 0, stream>>>(x_emb, adj1, s_d, xd2, adj2,
                                                   &lraw[1], &eraw[1], 8, 4, EMB_D);

    // ---- level 2: dense, n=4 -> C=2 ----
    dense_stack(xd2, adj2, 4, EMB_D, POOL_D, 2, POOL_BASE[2], s_raw, 0);
    dense_stack(xd2, adj2, 4, EMB_D, EMB_D, EMB_D, EMB_BASE[2], x_emb, 1);
    softmax_rows_kernel<<<blocks(B_G * 4, 256), 256, 0, stream>>>(s_raw, s_d, B_G * 4, 2);
    diffpool_small_kernel<<<B_G, 128, 0, stream>>>(x_emb, adj2, s_d, xd3, adj3,
                                                   &lraw[2], &eraw[2], 4, 2, EMB_D);

    // ---- head ----
    finalize_kernel<<<1, 64, 0, stream>>>(xd3, F(MLP_W), F(MLP_B), lraw, eraw, (float*)d_out);
}
